// GraphAttentionLayer_14637248544850
// MI455X (gfx1250) — matrix operand order repaired and checked by reading, not compiled, not run on hardware
//
#include <hip/hip_runtime.h>
#include <hip/hip_bf16.h>

#define N_NODES 8192
#define IN_F    256
#define OUT_F   128
#define ALPHA   0.2f

typedef __attribute__((ext_vector_type(16))) __bf16 v16bf;
typedef __attribute__((ext_vector_type(8)))  __bf16 v8bf;
typedef __attribute__((ext_vector_type(8)))  float  v8f;
typedef __attribute__((address_space(3)))    int    lds_int_t;

// ---------------------------------------------------------------------------
// 1) h (fp32, row-major [N, IN_F]) -> bf16
// ---------------------------------------------------------------------------
__global__ void cvt_h_bf16(const float* __restrict__ h, __bf16* __restrict__ hb, int n) {
    int i = blockIdx.x * blockDim.x + threadIdx.x;
    if (i < n) hb[i] = (__bf16)h[i];
}

// ---------------------------------------------------------------------------
// 2) W [IN_F, OUT_F] fp32 -> packed bf16 B-fragments for wmma_f32_16x16x32_bf16
//    Wp[kt][nt][lane][e]: lane l holds column N = nt*16 + (l&15),
//    K = kt*32 + 8*(l>>4) + (e&7) + 16*(e>>3)  (e = 0..15).
// ---------------------------------------------------------------------------
__global__ void pack_W_bf16(const float* __restrict__ W, __bf16* __restrict__ Wp) {
    int idx = blockIdx.x * blockDim.x + threadIdx.x;   // 8*8*32*16 = 32768
    int e    = idx & 15;
    int lane = (idx >> 4) & 31;
    int nt   = (idx >> 9) & 7;
    int kt   = idx >> 12;
    int k = kt * 32 + 8 * (lane >> 4) + (e & 7) + 16 * (e >> 3);
    int n = nt * 16 + (lane & 15);
    Wp[idx] = (__bf16)W[(size_t)k * OUT_F + n];
}

// ---------------------------------------------------------------------------
// 3) Wh = h @ W  (bf16 WMMA, f32 accumulate). One wave per 16-row tile.
//    Epilogue writes Wh fp32 row-major AND Whp in packed B-fragment layout.
// ---------------------------------------------------------------------------
__global__ void gemm_wh(const __bf16* __restrict__ hb, const __bf16* __restrict__ Wp,
                        float* __restrict__ Wh, __bf16* __restrict__ Whp) {
    int lane = threadIdx.x & 31;
    int wave = threadIdx.x >> 5;
    int mt   = blockIdx.x * 4 + wave;     // 0..511
    int half = lane >> 4;
    int rl   = lane & 15;
    int r    = mt * 16 + rl;

    v8f acc[8];
    #pragma unroll
    for (int nt = 0; nt < 8; ++nt)
        acc[nt] = (v8f){0.f, 0.f, 0.f, 0.f, 0.f, 0.f, 0.f, 0.f};

    #pragma unroll
    for (int kt = 0; kt < IN_F / 32; ++kt) {
        const __bf16* ap = hb + (size_t)r * IN_F + kt * 32 + 8 * half;
        v8bf a0 = *(const v8bf*)ap;
        v8bf a1 = *(const v8bf*)(ap + 16);
        v16bf A;
        #pragma unroll
        for (int i = 0; i < 8; ++i) { A[i] = a0[i]; A[i + 8] = a1[i]; }

        #pragma unroll
        for (int nt = 0; nt < 8; ++nt) {
            v16bf B = *(const v16bf*)(Wp + ((size_t)((kt * 8 + nt) * 32 + lane) << 4));
            acc[nt] = __builtin_amdgcn_wmma_f32_16x16x32_bf16(
                false, A, false, B, (short)0, acc[nt], false, false);
        }
    }

    int kt2 = mt >> 1;
    int lp  = (half << 4) | rl;
    #pragma unroll
    for (int nt = 0; nt < 8; ++nt) {
        #pragma unroll
        for (int v = 0; v < 8; ++v) {
            int row = mt * 16 + v + 8 * half;
            float x = acc[nt][v];
            Wh[(size_t)row * OUT_F + nt * 16 + rl] = x;
            Whp[((size_t)((kt2 * 8 + nt) * 32 + lp) << 4) + 8 * (mt & 1) + v] = (__bf16)x;
        }
    }
}

// ---------------------------------------------------------------------------
// 4) src = Wh @ a1, dst = Wh @ a2
// ---------------------------------------------------------------------------
__global__ void src_dst_k(const float* __restrict__ Wh, const float* __restrict__ a,
                          float* __restrict__ src, float* __restrict__ dst) {
    int n = blockIdx.x * blockDim.x + threadIdx.x;   // 0..8191
    const float* row = Wh + (size_t)n * OUT_F;
    float s = 0.f, d = 0.f;
    #pragma unroll 8
    for (int i = 0; i < OUT_F; ++i) {
        float w = row[i];
        s += w * a[i];
        d += w * a[OUT_F + i];
    }
    src[n] = s;
    dst[n] = d;
}

// ---------------------------------------------------------------------------
// 5) Flash-fused masked softmax + attention @ Wh + ELU, split-K + async ring.
//    One workgroup (4 waves) per 16-row block; wave w streams the contiguous
//    column quarter [w*2048, (w+1)*2048). adj tiles are staged through a
//    4-deep LDS ring buffer with GLOBAL_LOAD_ASYNC_TO_LDS_B128 / ASYNCcnt,
//    so fetch runs 4 tiles ahead of the exp/softmax VALU work and WMMAs.
// ---------------------------------------------------------------------------
#define ADEPTH 4

__global__ void gat_flash(const float* __restrict__ src, const float* __restrict__ dst,
                          const int* __restrict__ adj, const __bf16* __restrict__ Whp,
                          float* __restrict__ out) {
    __shared__ int   adj_tile[4][ADEPTH][16 * 32];   // 32 KB: per-wave ring
    __shared__ float lds_m[4][16];
    __shared__ float lds_l[4][16];
    __shared__ float lds_acc[4][16][OUT_F];          // 32 KB

    int lane = threadIdx.x & 31;
    int wave = threadIdx.x >> 5;          // 0..3 : K quarter
    int blk  = blockIdx.x;                // 0..511 : 16-row block
    int half = lane >> 4;
    int rl   = lane & 15;
    int r    = blk * 16 + rl;

    float m = -1e30f, lsum = 0.f;
    v8f acc[8];
    #pragma unroll
    for (int nt = 0; nt < 8; ++nt)
        acc[nt] = (v8f){0.f, 0.f, 0.f, 0.f, 0.f, 0.f, 0.f, 0.f};

    float s = src[r];

    const int KT = (N_NODES / 32) / 4;    // 64 K-tiles per wave
    const int k0 = wave * KT;

    // --- async-copy setup ------------------------------------------------
    // Tile = 16 rows x 32 cols of adj (2 KB). 4 x b128 per tile; per-lane
    // flat element idx = j*128 + lane*4 -> row = idx>>5, col = idx&31.
    lds_int_t* ltile0 = (lds_int_t*)&adj_tile[wave][0][0];
    unsigned long long gbase[4];
    #pragma unroll
    for (int j = 0; j < 4; ++j) {
        int idx = j * 128 + lane * 4;
        int row = idx >> 5, col = idx & 31;
        gbase[j] = (unsigned long long)(uintptr_t)
                   (adj + (size_t)(blk * 16 + row) * N_NODES + k0 * 32 + col);
    }

    // prologue: fill the ring with tiles k0 .. k0+ADEPTH-1
    #pragma unroll
    for (int d = 0; d < ADEPTH; ++d) {
        #pragma unroll
        for (int j = 0; j < 4; ++j) {
            asm volatile("global_load_async_to_lds_b128 %0, %1, off"
                         :: "v"(ltile0 + d * 512 + j * 128 + lane * 4),
                            "v"(gbase[j] + (unsigned long long)d * 128)
                         : "memory");
        }
    }

    for (int t = 0; t < KT; ++t) {
        int kt = k0 + t;

        // oldest tile resident: <= (ADEPTH-1)*4 async ops still outstanding
        if (t + ADEPTH - 1 < KT) {
            asm volatile("s_wait_asynccnt 12" ::: "memory");
        } else {
            asm volatile("s_wait_asynccnt 0" ::: "memory");
        }

        const int* tile = &adj_tile[wave][t & (ADEPTH - 1)][0];
        int4 ta0 = *(const int4*)(tile + rl * 32 + 8 * half);
        int4 ta1 = *(const int4*)(tile + rl * 32 + 8 * half + 4);
        int4 ta2 = *(const int4*)(tile + rl * 32 + 8 * half + 16);
        int4 ta3 = *(const int4*)(tile + rl * 32 + 8 * half + 20);

        int c0 = kt * 32 + 8 * half, c1 = c0 + 16;
        float4 cd0 = *(const float4*)(dst + c0);
        float4 cd1 = *(const float4*)(dst + c0 + 4);
        float4 cd2 = *(const float4*)(dst + c1);
        float4 cd3 = *(const float4*)(dst + c1 + 4);

        int   av[16] = {ta0.x, ta0.y, ta0.z, ta0.w, ta1.x, ta1.y, ta1.z, ta1.w,
                        ta2.x, ta2.y, ta2.z, ta2.w, ta3.x, ta3.y, ta3.z, ta3.w};
        float dv[16] = {cd0.x, cd0.y, cd0.z, cd0.w, cd1.x, cd1.y, cd1.z, cd1.w,
                        cd2.x, cd2.y, cd2.z, cd2.w, cd3.x, cd3.y, cd3.z, cd3.w};

        float lr[16];
        int msk = 0;
        float tmax = -1e30f;
        #pragma unroll
        for (int i = 0; i < 16; ++i) {
            float e  = s + dv[i];
            float le = e > 0.f ? e : ALPHA * e;     // leaky_relu(0.2)
            lr[i] = le;
            if (av[i] > 0) { msk |= (1 << i); tmax = fmaxf(tmax, le); }
        }
        tmax = fmaxf(tmax, __shfl_xor(tmax, 16, 32));
        float mn    = fmaxf(m, tmax);
        float scale = __expf(m - mn);

        float psum = 0.f;
        v16bf P;
        #pragma unroll
        for (int i = 0; i < 16; ++i) {
            float p = ((msk >> i) & 1) ? __expf(lr[i] - mn) : 0.f;
            psum += p;
            P[i] = (__bf16)p;
        }
        psum += __shfl_xor(psum, 16, 32);
        lsum = lsum * scale + psum;

        // rescale only when some row's max actually moved (wave-uniform)
        if (__any(mn > m)) {
            #pragma unroll
            for (int v = 0; v < 8; ++v) {
                float sv = __shfl(scale, v + 8 * half, 32);
                #pragma unroll
                for (int nt = 0; nt < 8; ++nt) acc[nt][v] *= sv;
            }
        }
        m = mn;

        #pragma unroll
        for (int nt = 0; nt < 8; ++nt) {
            v16bf B = *(const v16bf*)(Whp + ((size_t)((kt * 8 + nt) * 32 + lane) << 4));
            acc[nt] = __builtin_amdgcn_wmma_f32_16x16x32_bf16(
                false, P, false, B, (short)0, acc[nt], false, false);
        }

        // refill the ring slot we just consumed with tile t+ADEPTH
        if (t + ADEPTH < KT) {
            #pragma unroll
            for (int j = 0; j < 4; ++j) {
                asm volatile("global_load_async_to_lds_b128 %0, %1, off"
                             :: "v"(ltile0 + (t & (ADEPTH - 1)) * 512 + j * 128 + lane * 4),
                                "v"(gbase[j] + (unsigned long long)(t + ADEPTH) * 128)
                             : "memory");
            }
        }
    }

    // --- merge the 4 waves' online-softmax states through LDS ---
    if (lane < 16) { lds_m[wave][rl] = m; lds_l[wave][rl] = lsum; }
    __syncthreads();

    float mstar = lds_m[0][rl];
    #pragma unroll
    for (int w = 1; w < 4; ++w) mstar = fmaxf(mstar, lds_m[w][rl]);
    float factor = __expf(m - mstar);      // this wave's correction per row rl

    #pragma unroll
    for (int v = 0; v < 8; ++v) {
        float fv = __shfl(factor, v + 8 * half, 32);
        #pragma unroll
        for (int nt = 0; nt < 8; ++nt)
            lds_acc[wave][v + 8 * half][nt * 16 + rl] = acc[nt][v] * fv;
    }
    __syncthreads();

    // --- final reduce + normalize + ELU: 128 threads over 16x128 outputs ---
    int tid = threadIdx.x;
    int row = tid >> 3;                    // 0..15
    int f0  = (tid & 7) * 16;              // 0,16,...,112
    float mst = lds_m[0][row];
    #pragma unroll
    for (int w = 1; w < 4; ++w) mst = fmaxf(mst, lds_m[w][row]);
    float lst = 0.f;
    #pragma unroll
    for (int w = 0; w < 4; ++w) lst += lds_l[w][row] * __expf(lds_m[w][row] - mst);
    float inv = lst > 0.f ? 1.f / lst : 0.f;

    float* orow = out + (size_t)(blk * 16 + row) * OUT_F + f0;
    #pragma unroll
    for (int j = 0; j < 16; ++j) {
        float sum = lds_acc[0][row][f0 + j] + lds_acc[1][row][f0 + j]
                  + lds_acc[2][row][f0 + j] + lds_acc[3][row][f0 + j];
        float x = sum * inv;
        float y = x > 0.f ? x : (__expf(x) - 1.f);   // elu, alpha=1
        orow[j] = y;
    }
}

// ---------------------------------------------------------------------------
extern "C" void kernel_launch(void* const* d_in, const int* in_sizes, int n_in,
                              void* d_out, int out_size, void* d_ws, size_t ws_size,
                              hipStream_t stream) {
    const float* h   = (const float*)d_in[0];   // [8192, 256]
    const int*   adj = (const int*)d_in[1];     // [8192, 8192]
    const float* W   = (const float*)d_in[2];   // [256, 128]
    const float* a   = (const float*)d_in[3];   // [256, 1]
    float* out = (float*)d_out;                 // [8192, 128]

    char* ws = (char*)d_ws;
    size_t off = 0;
    __bf16* hb  = (__bf16*)(ws + off); off += (size_t)N_NODES * IN_F * sizeof(__bf16);   // 4 MB
    __bf16* Wp  = (__bf16*)(ws + off); off += (size_t)8 * 8 * 32 * 16 * sizeof(__bf16);  // 64 KB
    float*  Wh  = (float*) (ws + off); off += (size_t)N_NODES * OUT_F * sizeof(float);   // 4 MB
    __bf16* Whp = (__bf16*)(ws + off); off += (size_t)N_NODES * OUT_F * sizeof(__bf16);  // 2 MB
    float*  src = (float*) (ws + off); off += (size_t)N_NODES * sizeof(float);
    float*  dst = (float*) (ws + off); off += (size_t)N_NODES * sizeof(float);

    int nh = N_NODES * IN_F;
    cvt_h_bf16 <<<(nh + 255) / 256, 256, 0, stream>>>(h, hb, nh);
    pack_W_bf16<<<(8 * 8 * 32 * 16) / 256, 256, 0, stream>>>(W, Wp);
    gemm_wh    <<<N_NODES / 16 / 4, 128, 0, stream>>>(hb, Wp, Wh, Whp);
    src_dst_k  <<<N_NODES / 256, 256, 0, stream>>>(Wh, a, src, dst);
    gat_flash  <<<N_NODES / 16, 128, 0, stream>>>(src, dst, adj, Whp, out);
}
